// bLSTM_76089640616538
// MI455X (gfx1250) — matrix operand order repaired and checked by reading, not compiled, not run on hardware
//
#include <hip/hip_runtime.h>
#include <hip/hip_bf16.h>
#include <math.h>

// ---------------------------------------------------------------------------
// LSTM on MI455X (gfx1250): f16 WMMA GEMMs with f32 accumulation.
// B=64, T=256, I=512, H=1024, O=256
// ---------------------------------------------------------------------------

#define B_  64
#define T_  256
#define I_  512
#define H_  1024
#define O_  256
#define G4H_ 4096   // 4*H

typedef _Float16 half_t;
typedef __attribute__((ext_vector_type(8)))  half_t v8h;
typedef __attribute__((ext_vector_type(16))) half_t v16h;
typedef __attribute__((ext_vector_type(8)))  float  v8f;

// --- optional async global->LDS staging (gfx1250 ASYNCcnt path) -------------
#if defined(__has_builtin)
#if __has_builtin(__builtin_amdgcn_global_load_async_to_lds_b128) && \
    __has_builtin(__builtin_amdgcn_s_wait_asynccnt)
#define USE_ASYNC_LDS 1
#endif
#endif

#if defined(USE_ASYNC_LDS)
typedef int v4i_t __attribute__((vector_size(16)));
typedef __attribute__((address_space(1))) v4i_t gv4i;   // global int4
typedef __attribute__((address_space(3))) v4i_t sv4i;   // LDS int4
#endif

// copy 16 bytes (8 halfs) global -> LDS
__device__ __forceinline__ void ldscopy16(half_t* dst, const half_t* src) {
#if defined(USE_ASYNC_LDS)
    __builtin_amdgcn_global_load_async_to_lds_b128(
        (gv4i*)(void*)src, (sv4i*)(void*)dst, 0, 0);
#else
    *(v8h*)dst = *(const v8h*)src;
#endif
}

// make all issued LDS copies visible (call before __syncthreads())
__device__ __forceinline__ void ldscopy_fence() {
#if defined(USE_ASYNC_LDS)
    __builtin_amdgcn_s_wait_asynccnt(0);
#endif
}

__device__ __forceinline__ v16h mk16(v8h lo, v8h hi) {
    return __builtin_shufflevector(lo, hi, 0,1,2,3,4,5,6,7,8,9,10,11,12,13,14,15);
}

// A-fragment (16x32 f16, M x K). ISA layout:
//  lanes 0-15 : M=lane,    elements hold K = {0..7, 16..23}
//  lanes 16-31: M=lane-16, elements hold K = {8..15, 24..31}
// LDS tile layout: row-major [rows][32], 64-byte row stride.
__device__ __forceinline__ v16h frag_a(const half_t* sA, int mbase, int lane) {
    const half_t* p = sA + (mbase + (lane & 15)) * 32 + ((lane & 16) ? 8 : 0);
    v8h lo = *(const v8h*)(p);
    v8h hi = *(const v8h*)(p + 16);
    return mk16(lo, hi);
}

// B-fragment (32x16 f16, K x N). ISA layout:
//  lanes 0-15 : N=lane,    elements hold K = {0..7, 16..23}
//  lanes 16-31: N=lane-16, elements hold K = {8..15, 24..31}
// LDS tile layout: column-major [cols][32] (each col's 32 K-values contiguous).
__device__ __forceinline__ v16h frag_b(const half_t* sBcols, int lane) {
    const half_t* p = sBcols + (lane & 15) * 32 + ((lane & 16) ? 8 : 0);
    v8h lo = *(const v8h*)(p);
    v8h hi = *(const v8h*)(p + 16);
    return mk16(lo, hi);
}

__device__ __forceinline__ v8f wmma_f16(v16h a, v16h b, v8f c) {
    return __builtin_amdgcn_wmma_f32_16x16x32_f16(
        /*neg_a=*/false, a, /*neg_b=*/false, b,
        /*c_mod=*/(short)0, c, /*reuse_a=*/false, /*reuse_b=*/false);
}

__device__ __forceinline__ float sigmoid_f(float x) {
    return 1.0f / (1.0f + __expf(-x));
}

// ---------------------------------------------------------------------------
// f32 -> f16 cast (grid-stride)
// ---------------------------------------------------------------------------
__global__ void kcvt(const float* __restrict__ s, half_t* __restrict__ d, int n) {
    for (int i = blockIdx.x * blockDim.x + threadIdx.x; i < n;
         i += gridDim.x * blockDim.x)
        d[i] = (half_t)s[i];
}

// zero h0 (f16) and c0 (f32)
__global__ void kinit(half_t* __restrict__ h, float* __restrict__ c, int n) {
    int i = blockIdx.x * blockDim.x + threadIdx.x;
    if (i < n) { h[i] = (half_t)0.0f; c[i] = 0.0f; }
}

// ---------------------------------------------------------------------------
// x_gates = seq @ W_ih^T + (b_ih + b_hh)    [M=16384, N=4096, K=512] -> f16
// Block: 128 threads (4 waves), 64x64 tile. Wave w: M-tile w, 4 N-tiles.
// ---------------------------------------------------------------------------
__global__ __launch_bounds__(128) void kgemm_xg(
    const half_t* __restrict__ A,     // [16384, 512] f16 (seq)
    const half_t* __restrict__ Wih,   // [4096, 512]  f16
    const float*  __restrict__ b_ih,
    const float*  __restrict__ b_hh,
    half_t* __restrict__ xg)          // [16384, 4096] f16
{
    __shared__ __align__(16) half_t sA[64 * 32];
    __shared__ __align__(16) half_t sB[64 * 32];

    const int tid  = threadIdx.x;
    const int lane = tid & 31;
    const int w    = tid >> 5;
    const int M0   = blockIdx.x * 64;
    const int N0   = blockIdx.y * 64;

    v8f zero = 0.0f;
    v8f acc[4] = { zero, zero, zero, zero };

    for (int k0 = 0; k0 < I_; k0 += 32) {
        // stage A (64x32) and B (64 cols x 32 k): each thread copies 16 halfs of each
        const int r    = tid >> 1;
        const int part = (tid & 1) * 16;
        ldscopy16(&sA[r * 32 + part],     &A[(size_t)(M0 + r) * I_ + k0 + part]);
        ldscopy16(&sA[r * 32 + part + 8], &A[(size_t)(M0 + r) * I_ + k0 + part + 8]);
        ldscopy16(&sB[r * 32 + part],     &Wih[(size_t)(N0 + r) * I_ + k0 + part]);
        ldscopy16(&sB[r * 32 + part + 8], &Wih[(size_t)(N0 + r) * I_ + k0 + part + 8]);
        if (k0 + 32 < I_)
            __builtin_prefetch(&A[(size_t)(M0 + r) * I_ + k0 + 32], 0, 1);
        ldscopy_fence();
        __syncthreads();

        v16h a = frag_a(sA, w * 16, lane);
        #pragma unroll
        for (int n = 0; n < 4; ++n) {
            v16h b = frag_b(&sB[(n * 16) * 32], lane);
            acc[n] = wmma_f16(a, b, acc[n]);
        }
        __syncthreads();
    }

    const int rowoff = (lane & 16) ? 8 : 0;
    const int nlane  = lane & 15;
    #pragma unroll
    for (int n = 0; n < 4; ++n) {
        const int col  = N0 + n * 16 + nlane;
        const float bias = b_ih[col] + b_hh[col];
        #pragma unroll
        for (int r = 0; r < 8; ++r) {
            const int m = M0 + w * 16 + r + rowoff;
            xg[(size_t)m * G4H_ + col] = (half_t)(acc[n][r] + bias);
        }
    }
}

// ---------------------------------------------------------------------------
// One recurrent step (launched T times):
//   gates = x_gates[:,t,:] + h_prev @ W_hh^T ; cell/hidden update fused.
// Grid: 16 blocks, each owns 64 H-columns (for all 4 gates).
// Block: 256 threads = 8 waves. Wave w: M-tile (w&3), column-half (w>>2).
// Accumulators acc[gate][ntile] -> i/f/g/o for a (b,h) pair share a lane slot.
// ---------------------------------------------------------------------------
__global__ __launch_bounds__(256) void klstm_step(
    const half_t* __restrict__ xg,      // [B, T, 4H] f16
    const half_t* __restrict__ h_prev,  // [B, H] f16
    const half_t* __restrict__ Whh,     // [4H, H] f16
    float*        __restrict__ c,       // [B, H] f32 (in/out)
    half_t*       __restrict__ h_out,   // [B, H] f16
    int t)
{
    __shared__ __align__(16) half_t sA[64 * 32];    // h chunk   (4 KB)
    __shared__ __align__(16) half_t sB[256 * 32];   // Whh chunk (16 KB)

    const int tid  = threadIdx.x;
    const int lane = tid & 31;
    const int w    = tid >> 5;
    const int mt   = w & 3;      // batch tile: rows mt*16 .. mt*16+15
    const int ch   = w >> 2;     // column half: 32 cols
    const int N0   = blockIdx.x * 64;

    v8f zero = 0.0f;
    v8f acc[4][2];
    #pragma unroll
    for (int g = 0; g < 4; ++g) { acc[g][0] = zero; acc[g][1] = zero; }

    for (int k0 = 0; k0 < H_; k0 += 32) {
        // stage A: h_prev[0..63, k0..k0+31]
        {
            const int r    = tid >> 2;
            const int part = (tid & 3) * 8;
            ldscopy16(&sA[r * 32 + part], &h_prev[(size_t)r * H_ + k0 + part]);
        }
        // stage B: 4 gates x 64 cols x 32 k ; sB row index gc = gate*64 + cc
        {
            const int gate = tid >> 6;
            const int cc   = tid & 63;
            const half_t* src = Whh + (size_t)(gate * H_ + N0 + cc) * H_ + k0;
            half_t* dst = &sB[tid * 32];
            ldscopy16(dst + 0,  src + 0);
            ldscopy16(dst + 8,  src + 8);
            ldscopy16(dst + 16, src + 16);
            ldscopy16(dst + 24, src + 24);
            if (k0 + 32 < H_)
                __builtin_prefetch(src + 32, 0, 1);
        }
        ldscopy_fence();
        __syncthreads();

        v16h a = frag_a(sA, mt * 16, lane);
        #pragma unroll
        for (int g = 0; g < 4; ++g) {
            #pragma unroll
            for (int n = 0; n < 2; ++n) {
                v16h b = frag_b(&sB[(g * 64 + ch * 32 + n * 16) * 32], lane);
                acc[g][n] = wmma_f16(a, b, acc[g][n]);
            }
        }
        __syncthreads();
    }

    // fused gate nonlinearity + cell/hidden update
    const int rowoff = (lane & 16) ? 8 : 0;
    const int nlane  = lane & 15;
    #pragma unroll
    for (int n = 0; n < 2; ++n) {
        const int col = N0 + ch * 32 + n * 16 + nlane;
        #pragma unroll
        for (int r = 0; r < 8; ++r) {
            const int b = mt * 16 + r + rowoff;
            const size_t xbase = ((size_t)b * T_ + t) * G4H_ + col;
            const float xi = acc[0][n][r] + (float)xg[xbase];
            const float xf = acc[1][n][r] + (float)xg[xbase + H_];
            const float xc = acc[2][n][r] + (float)xg[xbase + 2 * H_];
            const float xo = acc[3][n][r] + (float)xg[xbase + 3 * H_];
            const float ig = sigmoid_f(xi);
            const float fg = sigmoid_f(xf);
            const float gg = tanhf(xc);
            const float og = sigmoid_f(xo);
            const int idx = b * H_ + col;
            const float cn = fg * c[idx] + ig * gg;
            c[idx]     = cn;
            h_out[idx] = (half_t)(og * tanhf(cn));
        }
    }
}

// ---------------------------------------------------------------------------
// Output projection + log-softmax for one timestep.
// Grid: 4 blocks x 16 batch rows; block 128 threads (4 waves) covers all
// 256 output columns -> row-wise log-softmax entirely in LDS.
// ---------------------------------------------------------------------------
__global__ __launch_bounds__(128) void kout_proj(
    const half_t* __restrict__ h,      // [B, H] f16
    const half_t* __restrict__ Wout,   // [O, H] f16
    const float*  __restrict__ b_out,  // [O]
    float*        __restrict__ out,    // [B, T, O] f32
    int t)
{
    __shared__ __align__(16) half_t sA[16 * 32];
    __shared__ __align__(16) half_t sB[256 * 32];
    __shared__ float slog[16 * 256];
    __shared__ float sden[16];

    const int tid  = threadIdx.x;
    const int lane = tid & 31;
    const int w    = tid >> 5;
    const int M0   = blockIdx.x * 16;

    v8f zero = 0.0f;
    v8f acc[4] = { zero, zero, zero, zero };

    for (int k0 = 0; k0 < H_; k0 += 32) {
        if (tid < 64) {
            const int r    = tid >> 2;
            const int part = (tid & 3) * 8;
            ldscopy16(&sA[r * 32 + part], &h[(size_t)(M0 + r) * H_ + k0 + part]);
        }
        for (int cc = tid; cc < 256; cc += 128) {
            const half_t* src = Wout + (size_t)cc * H_ + k0;
            half_t* dst = &sB[cc * 32];
            ldscopy16(dst + 0,  src + 0);
            ldscopy16(dst + 8,  src + 8);
            ldscopy16(dst + 16, src + 16);
            ldscopy16(dst + 24, src + 24);
        }
        ldscopy_fence();
        __syncthreads();

        v16h a = frag_a(sA, 0, lane);
        #pragma unroll
        for (int n = 0; n < 4; ++n) {
            v16h b = frag_b(&sB[(w * 64 + n * 16) * 32], lane);
            acc[n] = wmma_f16(a, b, acc[n]);
        }
        __syncthreads();
    }

    const int rowoff = (lane & 16) ? 8 : 0;
    const int nlane  = lane & 15;
    #pragma unroll
    for (int n = 0; n < 4; ++n) {
        const int col = w * 64 + n * 16 + nlane;
        #pragma unroll
        for (int r = 0; r < 8; ++r) {
            const int row = r + rowoff;
            slog[row * 256 + col] = acc[n][r] + b_out[col];
        }
    }
    __syncthreads();

    if (tid < 16) {
        const float* rowp = &slog[tid * 256];
        float mx = -INFINITY;
        for (int j = 0; j < 256; ++j) mx = fmaxf(mx, rowp[j]);
        float s = 0.0f;
        for (int j = 0; j < 256; ++j) s += __expf(rowp[j] - mx);
        sden[tid] = mx + __logf(s);
    }
    __syncthreads();

    for (int i = tid; i < 16 * 256; i += 128) {
        const int row = i >> 8;
        const int col = i & 255;
        out[((size_t)(M0 + row) * T_ + t) * O_ + col] = slog[i] - sden[row];
    }
}

// ---------------------------------------------------------------------------
// host launcher
// ---------------------------------------------------------------------------
extern "C" void kernel_launch(void* const* d_in, const int* in_sizes, int n_in,
                              void* d_out, int out_size, void* d_ws, size_t ws_size,
                              hipStream_t stream) {
    const float* seq   = (const float*)d_in[0];   // [B,T,I]
    const float* W_ih  = (const float*)d_in[1];   // [4H,I]
    const float* W_hh  = (const float*)d_in[2];   // [4H,H]
    const float* b_ih  = (const float*)d_in[3];   // [4H]
    const float* b_hh  = (const float*)d_in[4];   // [4H]
    const float* W_out = (const float*)d_in[5];   // [O,H]
    const float* b_out = (const float*)d_in[6];   // [O]
    float* out = (float*)d_out;                   // [B,T,O]

    // workspace carve-up (256B aligned)
    char* ws = (char*)d_ws;
    size_t off = 0;
    auto carve = [&](size_t bytes) -> void* {
        void* p = ws + off;
        off = (off + bytes + 255) & ~(size_t)255;
        return p;
    };
    half_t* seq_h  = (half_t*)carve((size_t)B_ * T_ * I_ * 2);
    half_t* wih_h  = (half_t*)carve((size_t)G4H_ * I_ * 2);
    half_t* whh_h  = (half_t*)carve((size_t)G4H_ * H_ * 2);
    half_t* wout_h = (half_t*)carve((size_t)O_ * H_ * 2);
    half_t* xg     = (half_t*)carve((size_t)B_ * T_ * G4H_ * 2);
    half_t* h_a    = (half_t*)carve((size_t)B_ * H_ * 2);
    half_t* h_b    = (half_t*)carve((size_t)B_ * H_ * 2);
    float*  cbuf   = (float*)carve((size_t)B_ * H_ * 4);
    (void)ws_size; (void)n_in; (void)in_sizes; (void)out_size;

    // 1) f32 -> f16 conversions
    kcvt<<<2048, 256, 0, stream>>>(seq,   seq_h,  B_ * T_ * I_);
    kcvt<<<1024, 256, 0, stream>>>(W_ih,  wih_h,  G4H_ * I_);
    kcvt<<<1024, 256, 0, stream>>>(W_hh,  whh_h,  G4H_ * H_);
    kcvt<<<256,  256, 0, stream>>>(W_out, wout_h, O_ * H_);

    // 2) zero h0, c0
    kinit<<<(B_ * H_ + 255) / 256, 256, 0, stream>>>(h_a, cbuf, B_ * H_);

    // 3) x_gates big GEMM
    {
        dim3 grid(B_ * T_ / 64, G4H_ / 64);   // 256 x 64
        kgemm_xg<<<grid, 128, 0, stream>>>(seq_h, wih_h, b_ih, b_hh, xg);
    }

    // 4) recurrence + per-step output projection
    for (int t = 0; t < T_; ++t) {
        half_t* hp = (t & 1) ? h_b : h_a;
        half_t* hn = (t & 1) ? h_a : h_b;
        klstm_step<<<H_ / 64, 256, 0, stream>>>(xg, hp, whh_h, cbuf, hn, t);
        kout_proj<<<B_ / 16, 128, 0, stream>>>(hn, wout_h, b_out, out, t);
    }
}
